// Model_55868934586770
// MI455X (gfx1250) — compile-verified, matrix operand order checked by reading, not run on hardware
//
#include <hip/hip_runtime.h>
#include <hip/hip_bf16.h>

// ---------------------------------------------------------------------------
// Fused courier/building MLP scorer for gfx1250 (MI455X), bf16 WMMA path.
//   rows r = b*2000 + n, M = 256,000
//   X[r, 0:64]    = emb_courier[cid[b]]
//   X[r, 64:128]  = emb_building[bid[b]]
//   X[r, 128:192] = emb_building[n]
//   X[r, 192:320] = ft[b, n, :]
//   y = (relu(relu(X W1^T + b1) W2^T + b2)) W3^T + b3 ; y[b, bid[b]] = -inf
//
// Round 3: weight nt-blocks staged into LDS ONCE per workgroup (8x L2 traffic
// cut vs per-wave streaming), double-buffered with global_load_async_to_lds_b128
// (ASYNCcnt) overlapping the WMMA loop. Single shared compute loop; the
// no-workspace fallback only changes the LDS fill (convert-fill vs async copy).
// ---------------------------------------------------------------------------

typedef __attribute__((ext_vector_type(16))) __bf16 v16bf;
typedef __attribute__((ext_vector_type(8)))  float  v8f;

#define NB      2000
#define DIM_IN  320
#define HD1     512
#define MTILE   128
#define KT1     10            // 320/32 k-tiles, layer 1
#define KT2     16            // 512/32 k-tiles, layer 2

// LDS layout -----------------------------------------------------------------
#define LDX_U32 164                      // X row stride in u32 (160 pairs + pad, %4==0)
#define LDH_U16 520                      // H1/H2 row stride in u16 (512 + pad; /2 %4==0)
#define SZ_H    (MTILE * LDH_U16 * 2)    // 133,120 B per hidden buffer
#define OFF_H1  SZ_H                     // 133,120
#define OFF_W3  (2 * SZ_H)               // 266,240 (2 KB)
#define OFF_B1  (OFF_W3 + 2048)          // 268,288 (2 KB)
#define OFF_B2  (OFF_B1 + 2048)          // 270,336 (2 KB)
#define OFF_PART (OFF_B2 + 2048)         // 272,384 (1 KB)
#define OFF_BW  (OFF_PART + 1024)        // 273,408
#define BW_STRIDE 16384                  // one nt-block (<= KT2*1024 B)
#define SMEM_BYTES (OFF_BW + 2 * BW_STRIDE)   // 306,176 B (< 320 KB/WGP)

// Workspace layout: pre-swizzled bf16 weights, fragment-major
//   tile t = nt*KT + kt -> 256 u32 ; within tile: lane*8 + j
#define WS_W1_U32 (32 * KT1 * 256)       // 327,680 B
#define WS_W2_U32 (32 * KT2 * 256)       // 524,288 B
#define WS_BYTES  ((WS_W1_U32 + WS_W2_U32) * 4)   // 851,968 B

// bf16 helpers ---------------------------------------------------------------
__device__ __forceinline__ unsigned short f2bf(float f) {
    unsigned int u = __float_as_uint(f);
    u += 0x7FFFu + ((u >> 16) & 1u);     // round-to-nearest-even
    return (unsigned short)(u >> 16);
}
__device__ __forceinline__ unsigned int packbf2(float lo, float hi) {
    return ((unsigned int)f2bf(hi) << 16) | (unsigned int)f2bf(lo);
}
__device__ __forceinline__ float bf2f(unsigned short h) {
    return __uint_as_float(((unsigned int)h) << 16);
}

union FragBF16 {
    unsigned int u[8];
    uint4        u4[2];
    v16bf        v;
};

// ---- CDNA5 async global->LDS (ASYNCcnt) ------------------------------------
// Shared aperture is 4GB-aligned -> low 32 bits of a generic LDS pointer are
// the LDS byte offset (see ISA 10.2).
__device__ __forceinline__ unsigned lds_off_of(const void* p) {
    return (unsigned)(unsigned long long)p;
}
__device__ __forceinline__ void async_copy_b128(unsigned lds_off, const void* gaddr) {
    asm volatile("global_load_async_to_lds_b128 %0, %1, off"
                 :: "v"(lds_off), "v"(gaddr) : "memory");
}
__device__ __forceinline__ void wait_async0() {
    asm volatile("s_wait_asynccnt 0x0" ::: "memory");
}

// A-fragment (16x32 bf16) from LDS, row-major packed u32 pairs, 2x ds_load_b128.
// ISA 16-bit A layout: lanes 0-15 -> M=0..15 {K 0..7,16..23}; lanes 16-31 -> {K 8..15,24..31}
__device__ __forceinline__ FragBF16
load_a_lds(const unsigned int* S, int mbase, int kbase, int ldu32, int lane) {
    FragBF16 f;
    const int m  = lane & 15;
    const int kh = lane >> 4;
    const uint4* rp = (const uint4*)(S + (mbase + m) * ldu32 + (kbase >> 1) + kh * 4);
    f.u4[0] = rp[0];      // kg=0
    f.u4[1] = rp[2];      // kg=1 (+8 u32)
    return f;
}

// B-fragment from LDS-staged fragment-major block: 2x ds_load_b128.
__device__ __forceinline__ v16bf
load_b_lds(const unsigned int* buf, int kt, int lane) {
    FragBF16 f;
    const uint4* p = (const uint4*)(buf + kt * 256 + lane * 8);
    f.u4[0] = p[0];
    f.u4[1] = p[1];
    return f.v;
}

// Fill one nt-block of weight fragments into LDS.
// preconv path: async b128 copies from pre-swizzled bf16 workspace.
__device__ __forceinline__ void
stage_wblk_async(const unsigned int* __restrict__ Wp, int blk, int KT,
                 void* buf, int tid) {
    const unsigned int* src = Wp + (size_t)blk * KT * 256;
    const unsigned base = lds_off_of(buf);
    const int nchunks = KT * 64;               // 16B chunks
#pragma unroll 1
    for (int c = tid; c < nchunks; c += 256)
        async_copy_b128(base + c * 16, (const void*)(src + c * 4));
}
// fallback path: cooperative fp32->bf16 convert-fill (once per block, not per wave).
__device__ __forceinline__ void
stage_wblk_f32(const float* __restrict__ W, int nt, int KT, int ldk,
               unsigned int* buf, int tid) {
#pragma unroll 1
    for (int c = tid; c < KT * 256; c += 256) {
        const int kt = c >> 8, idx = c & 255;
        const int lane = idx >> 3, j = idx & 7;
        const int n  = nt * 16 + (lane & 15);
        const int kh = lane >> 4, kg = j >> 2, kp = j & 3;
        const int k0 = kt * 32 + kg * 16 + kh * 8 + kp * 2;
        const float2 w = *(const float2*)(W + (size_t)n * ldk + k0);
        buf[c] = packbf2(w.x, w.y);
    }
}

// ---------------------------------------------------------------------------
// Pre-pass: convert W[N][K] fp32 -> bf16, swizzled into fragment order.
// ---------------------------------------------------------------------------
__global__ __launch_bounds__(256) void weight_swizzle_kernel(
    const float* __restrict__ W, unsigned int* __restrict__ dst,
    int KT, int ldk)
{
    const int t    = blockIdx.x;
    const int idx  = threadIdx.x;
    const int lane = idx >> 3;
    const int j    = idx & 7;
    const int nt   = t / KT;
    const int kt   = t - nt * KT;
    const int n    = nt * 16 + (lane & 15);
    const int kh   = lane >> 4;
    const int kg   = j >> 2, kp = j & 3;
    const int k0   = kt * 32 + kg * 16 + kh * 8 + kp * 2;
    const float2 w = *(const float2*)(W + (size_t)n * ldk + k0);
    dst[(size_t)t * 256 + idx] = packbf2(w.x, w.y);
}

// ---------------------------------------------------------------------------
// Main fused kernel.
// ---------------------------------------------------------------------------
__global__ __launch_bounds__(256) void fused_mlp_kernel(
    const int* __restrict__ cid, const int* __restrict__ bid,
    const float* __restrict__ ft,
    const float* __restrict__ embC, const float* __restrict__ embB,
    const float* __restrict__ W1, const float* __restrict__ b1,
    const float* __restrict__ W2, const float* __restrict__ b2,
    const float* __restrict__ W3, const float* __restrict__ b3,
    const unsigned int* __restrict__ W1p, const unsigned int* __restrict__ W2p,
    float* __restrict__ out)
{
    extern __shared__ char smem[];
    unsigned int*   sX    = (unsigned int*)smem;               // stage A/B (overlaps H2)
    unsigned short* sH2   = (unsigned short*)smem;             // stage C/D
    unsigned short* sH1   = (unsigned short*)(smem + OFF_H1);  // stage B/C
    float*          sW3   = (float*)(smem + OFF_W3);
    float*          sB1   = (float*)(smem + OFF_B1);
    float*          sB2   = (float*)(smem + OFF_B2);
    float*          sPart = (float*)(smem + OFF_PART);
    unsigned int*   bw0   = (unsigned int*)(smem + OFF_BW);
    unsigned int*   bw1   = (unsigned int*)(smem + OFF_BW + BW_STRIDE);

    const bool preconv = (W1p != nullptr);
    const int tid      = threadIdx.x;
    const int lane     = tid & 31;
    const int wave     = tid >> 5;                 // 8 waves
    const int row_base = blockIdx.x * MTILE;

    // ---- Stage A: gather X tile into LDS (bf16); preload W3/b1/b2 ---------
    sW3[tid]       = W3[tid];
    sW3[tid + 256] = W3[tid + 256];
    sB1[tid]       = b1[tid];
    sB1[tid + 256] = b1[tid + 256];
    sB2[tid]       = b2[tid];
    sB2[tid + 256] = b2[tid + 256];

#pragma unroll 1
    for (int rr = 0; rr < 16; ++rr) {
        const int row = wave * 16 + rr;
        const int r   = row_base + row;
        const int b   = r / NB;
        const int n   = r - b * NB;
        const float* pc = embC + (size_t)cid[b] * 64;
        const float* pb = embB + (size_t)bid[b] * 64;
        const float* pw = embB + (size_t)n * 64;
        const float* pf = ft   + (size_t)r * 128;
        unsigned int* dst = sX + row * LDX_U32;
#pragma unroll 1
        for (int pp = lane; pp < 160; pp += 32) {
            const int c = pp * 2;
            float2 v;
            if      (c < 64)  v = *(const float2*)(pc + c);
            else if (c < 128) v = *(const float2*)(pb + (c - 64));
            else if (c < 192) v = *(const float2*)(pw + (c - 128));
            else              v = *(const float2*)(pf + (c - 192));
            dst[pp] = packbf2(v.x, v.y);
        }
    }
    __syncthreads();

    // ---- Stage B: H1 = relu(X W1^T + b1), WMMA bf16, K=320 ----------------
    {
        const int mt = wave;                       // each wave owns one 16-row strip
        FragBF16 afr[KT1];
#pragma unroll
        for (int kt = 0; kt < KT1; ++kt)
            afr[kt] = load_a_lds(sX, mt * 16, kt * 32, LDX_U32, lane);

        if (preconv) { stage_wblk_async(W1p, 0, KT1, bw0, tid); wait_async0(); }
        else         { stage_wblk_f32(W1, 0, KT1, DIM_IN, bw0, tid); }
        __syncthreads();

#pragma unroll 1
        for (int nt = 0; nt < 32; ++nt) {
            unsigned int* cur = (nt & 1) ? bw1 : bw0;
            unsigned int* nxt = (nt & 1) ? bw0 : bw1;
            if (nt + 1 < 32) {
                if (preconv) stage_wblk_async(W1p, nt + 1, KT1, nxt, tid);
                else         stage_wblk_f32(W1, nt + 1, KT1, DIM_IN, nxt, tid);
            }
            v8f acc = {};
#pragma unroll
            for (int kt = 0; kt < KT1; ++kt) {
                v16bf bm = load_b_lds(cur, kt, lane);
                acc = __builtin_amdgcn_wmma_f32_16x16x32_bf16(
                          false, afr[kt].v, false, bm, (short)0, acc, false, false);
            }
            const int col  = nt * 16 + (lane & 15);
            const float bias = sB1[col];
            const int rb   = mt * 16 + ((lane >> 4) << 3);
#pragma unroll
            for (int j = 0; j < 8; ++j) {
                float x = fmaxf(acc[j] + bias, 0.0f);
                sH1[(rb + j) * LDH_U16 + col] = f2bf(x);
            }
            if (preconv && nt + 1 < 32) wait_async0();
            __syncthreads();
        }
    }

    // ---- Stage C: H2 = relu(H1 W2^T + b2), WMMA bf16, K=512 ---------------
    {
        const int mt = wave;
        const unsigned int* sH1u = (const unsigned int*)(smem + OFF_H1);
        FragBF16 afr[KT2];
#pragma unroll
        for (int kt = 0; kt < KT2; ++kt)
            afr[kt] = load_a_lds(sH1u, mt * 16, kt * 32, LDH_U16 / 2, lane);

        if (preconv) { stage_wblk_async(W2p, 0, KT2, bw0, tid); wait_async0(); }
        else         { stage_wblk_f32(W2, 0, KT2, HD1, bw0, tid); }
        __syncthreads();

#pragma unroll 1
        for (int nt = 0; nt < 32; ++nt) {
            unsigned int* cur = (nt & 1) ? bw1 : bw0;
            unsigned int* nxt = (nt & 1) ? bw0 : bw1;
            if (nt + 1 < 32) {
                if (preconv) stage_wblk_async(W2p, nt + 1, KT2, nxt, tid);
                else         stage_wblk_f32(W2, nt + 1, KT2, HD1, nxt, tid);
            }
            v8f acc = {};
#pragma unroll
            for (int kt = 0; kt < KT2; ++kt) {
                v16bf bm = load_b_lds(cur, kt, lane);
                acc = __builtin_amdgcn_wmma_f32_16x16x32_bf16(
                          false, afr[kt].v, false, bm, (short)0, acc, false, false);
            }
            const int col  = nt * 16 + (lane & 15);
            const float bias = sB2[col];
            const int rb   = mt * 16 + ((lane >> 4) << 3);
#pragma unroll
            for (int j = 0; j < 8; ++j) {
                float x = fmaxf(acc[j] + bias, 0.0f);
                sH2[(rb + j) * LDH_U16 + col] = f2bf(x);
            }
            if (preconv && nt + 1 < 32) wait_async0();
            __syncthreads();
        }
    }

    // ---- Stage D: y = H2 . w3 + b3 (GEMV), -inf fixup ---------------------
    {
        const int row  = tid >> 1;                 // 2 threads per row
        const int half = tid & 1;
        const unsigned short* hrow = sH2 + row * LDH_U16 + half * 256;
        const float* w = sW3 + half * 256;
        float s = 0.0f;
#pragma unroll 8
        for (int k = 0; k < 256; ++k)
            s += bf2f(hrow[k]) * w[k];
        sPart[tid] = s;
    }
    __syncthreads();
    if (tid < MTILE) {
        const int r = row_base + tid;
        const int b = r / NB;
        const int n = r - b * NB;
        float y = sPart[2 * tid] + sPart[2 * tid + 1] + b3[0];
        out[r] = (n == bid[b]) ? -__builtin_inff() : y;
    }
}

extern "C" void kernel_launch(void* const* d_in, const int* in_sizes, int n_in,
                              void* d_out, int out_size, void* d_ws, size_t ws_size,
                              hipStream_t stream) {
    const int*   cid  = (const int*)d_in[0];
    const int*   bid  = (const int*)d_in[1];
    const float* ft   = (const float*)d_in[2];
    const float* embC = (const float*)d_in[3];
    const float* embB = (const float*)d_in[4];
    const float* W1   = (const float*)d_in[5];
    const float* b1   = (const float*)d_in[6];
    const float* W2   = (const float*)d_in[7];
    const float* b2   = (const float*)d_in[8];
    const float* W3   = (const float*)d_in[9];
    const float* b3   = (const float*)d_in[10];
    float* out = (float*)d_out;

    dim3 grid(2000);     // 256,000 rows / 128-row M-tile
    dim3 block(256);     // 8 wave32

    if (ws_size >= (size_t)WS_BYTES) {
        unsigned int* W1p = (unsigned int*)d_ws;
        unsigned int* W2p = W1p + WS_W1_U32;
        weight_swizzle_kernel<<<dim3(32 * KT1), dim3(256), 0, stream>>>(W1, W1p, KT1, DIM_IN);
        weight_swizzle_kernel<<<dim3(32 * KT2), dim3(256), 0, stream>>>(W2, W2p, KT2, HD1);
        fused_mlp_kernel<<<grid, block, SMEM_BYTES, stream>>>(
            cid, bid, ft, embC, embB, W1, b1, W2, b2, W3, b3, W1p, W2p, out);
    } else {
        fused_mlp_kernel<<<grid, block, SMEM_BYTES, stream>>>(
            cid, bid, ft, embC, embB, W1, b1, W2, b2, W3, b3, nullptr, nullptr, out);
    }
}
